// RNNColorbot_28922309771872
// MI455X (gfx1250) — matrix operand order repaired and checked by reading, not compiled, not run on hardware
//
#include <hip/hip_runtime.h>
#include <hip/hip_bf16.h>

// ---------------------------------------------------------------------------
// RNNColorbot: 2-layer LSTM (B=512, T=128, D=256, H=1024) + dense head.
// bf16 WMMA (v_wmma_f32_16x16x32_bf16), fp32 accumulate, fused LSTM gates.
// Each wave: 32(M)x16(N) tile x 4 gates -> B fragments reused across 2 M-tiles.
// ---------------------------------------------------------------------------

typedef __attribute__((ext_vector_type(16))) __bf16 v16bf;
typedef __attribute__((ext_vector_type(8)))  __bf16 v8bf;
typedef __attribute__((ext_vector_type(8)))  float  v8f;

#define B_SZ  512
#define T_SZ  128
#define DIN   256
#define H_SZ  1024
#define G4    (4 * H_SZ)
#define K1CAT (DIN + H_SZ)      // layer-1 concat K = 1280
#define K2CAT (2 * H_SZ)        // layer-2 concat K = 2048

// Persistent device-side scratch (recomputed deterministically every call).
__device__ __attribute__((aligned(256))) __bf16 g_x[(size_t)B_SZ * T_SZ * DIN];     // bf16 copy of chars
__device__ __attribute__((aligned(256))) __bf16 g_Wt1[(size_t)G4 * K1CAT];          // [4H][K] = [W1;U1]^T
__device__ __attribute__((aligned(256))) __bf16 g_Wt2[(size_t)G4 * K2CAT];          // [4H][K] = [W2;U2]^T
__device__ __attribute__((aligned(256))) __bf16 g_h1[2][(size_t)B_SZ * H_SZ];       // ping-pong hidden L1
__device__ __attribute__((aligned(256))) __bf16 g_h2[2][(size_t)B_SZ * H_SZ];       // ping-pong hidden L2
__device__ __attribute__((aligned(256))) float  g_c1[(size_t)B_SZ * H_SZ];
__device__ __attribute__((aligned(256))) float  g_c2[(size_t)B_SZ * H_SZ];
__device__ __attribute__((aligned(256))) float  g_hsel[(size_t)B_SZ * H_SZ];        // h2 at t = seqlen-1

// ---------------------------------------------------------------------------
// Prep kernels
// ---------------------------------------------------------------------------
__global__ void k_cvt_x(const float* __restrict__ chars, int n) {
  int i = blockIdx.x * blockDim.x + threadIdx.x;
  if (i < n) g_x[i] = (__bf16)chars[i];
}

// g_Wt1[n][k] = (k < 256 ? W1[k][n] : U1[k-256][n]), bf16, row-major [4096][1280]
__global__ void k_pack_w1(const float* __restrict__ W1, const float* __restrict__ U1) {
  int i = blockIdx.x * blockDim.x + threadIdx.x;
  if (i >= G4 * K1CAT) return;
  int n = i / K1CAT, k = i % K1CAT;
  float v = (k < DIN) ? W1[(size_t)k * G4 + n] : U1[(size_t)(k - DIN) * G4 + n];
  g_Wt1[i] = (__bf16)v;
}

// g_Wt2[n][k] = (k < 1024 ? W2[k][n] : U2[k-1024][n]), bf16, row-major [4096][2048]
__global__ void k_pack_w2(const float* __restrict__ W2, const float* __restrict__ U2) {
  int i = blockIdx.x * blockDim.x + threadIdx.x;
  if (i >= G4 * K2CAT) return;
  int n = i / K2CAT, k = i % K2CAT;
  float v = (k < H_SZ) ? W2[(size_t)k * G4 + n] : U2[(size_t)(k - H_SZ) * G4 + n];
  g_Wt2[i] = (__bf16)v;
}

__global__ void k_zero_state() {
  int i = blockIdx.x * blockDim.x + threadIdx.x;
  if (i < B_SZ * H_SZ) {
    g_c1[i] = 0.f; g_c2[i] = 0.f;
    g_h1[0][i] = (__bf16)0.f; g_h2[0][i] = (__bf16)0.f;
  }
}

// ---------------------------------------------------------------------------
// WMMA fragment loaders (CDNA5 16-bit layouts, ISA 7.12.2)
// A 16x32: lane L<16 -> M=L, K chunks [k..k+7],[k+16..k+23];
//          lane L>=16 -> M=L-16, chunks [k+8..k+15],[k+24..k+31]
// ---------------------------------------------------------------------------
__device__ __forceinline__ v16bf load_a_frag(const __bf16* __restrict__ A, int ld,
                                             int m0, int k, int lane) {
  int m  = m0 + (lane & 15);
  int kb = k + ((lane >> 4) << 3);          // +8 for upper half-wave
  const __bf16* p = A + (size_t)m * ld + kb;
  union { v16bf v; struct { v8bf lo, hi; } s; } u;
  u.s.lo = *(const v8bf*)(p);               // K = kb .. kb+7
  u.s.hi = *(const v8bf*)(p + 16);          // K = kb+16 .. kb+23
  return u.v;
}

// B 32x16 from W^T stored [N][K]: lane L<16 -> N=L, K chunks [k..k+7],[k+16..k+23];
//                                 lane L>=16 -> N=L-16, chunks [k+8..k+15],[k+24..k+31]
__device__ __forceinline__ v16bf load_b_frag(const __bf16* __restrict__ Wt, int ld,
                                             int n0, int k, int lane) {
  int n  = n0 + (lane & 15);
  int kb = k + ((lane >> 4) << 3);          // +8 for upper half-wave
  const __bf16* p = Wt + (size_t)n * ld + kb;
  union { v16bf v; struct { v8bf lo, hi; } s; } u;
  u.s.lo = *(const v8bf*)(p);               // K = kb .. kb+7
  u.s.hi = *(const v8bf*)(p + 16);          // K = kb+16 .. kb+23
  return u.v;
}

__device__ __forceinline__ float sigf(float x) { return 1.f / (1.f + __expf(-x)); }

// ---------------------------------------------------------------------------
// Fused LSTM step: Z = [A0 | A1] @ Wcat + b for all 4 gates of a 32x16 tile,
// then gate math + state update, entirely in-register.
// Grid: (H/16, B/128), block 128 (4 waves); each wave owns a 32(M)x16(N) tile.
// ---------------------------------------------------------------------------
__global__ __launch_bounds__(128, 1)
void k_lstm_step(int layer, int t,
                 const float* __restrict__ bias,
                 const int* __restrict__ seqlen) {
  const int lane = threadIdx.x & 31;
  const int wave = threadIdx.x >> 5;
  const int n0   = blockIdx.x * 16;                       // hidden-column tile
  const int m0   = (blockIdx.y * 4 + wave) * 32;          // batch-row tile (2x16)

  const int rb = t & 1, wb = rb ^ 1;

  const __bf16* A0; int lda0, K0;
  const __bf16* A1;                                       // recurrent input, ld = H
  const __bf16* Wt; int ldk;
  float* cst; __bf16* hout; float* hsel;
  if (layer == 0) {
    A0 = g_x + (size_t)t * DIN; lda0 = T_SZ * DIN; K0 = DIN;
    A1 = g_h1[rb]; Wt = g_Wt1; ldk = K1CAT;
    cst = g_c1; hout = g_h1[wb]; hsel = nullptr;
  } else {
    A0 = g_h1[wb]; lda0 = H_SZ; K0 = H_SZ;                // freshly written h1(t)
    A1 = g_h2[rb]; Wt = g_Wt2; ldk = K2CAT;
    cst = g_c2; hout = g_h2[wb]; hsel = g_hsel;
  }
  const int kcat = K0 + H_SZ;

  // Accumulators: [m-subtile][gate] 16x16 f32 tiles, seeded with bias.
  v8f acc[2][4];
#pragma unroll
  for (int g = 0; g < 4; ++g) {
    float bi = bias[g * H_SZ + n0 + (lane & 15)];
#pragma unroll
    for (int r = 0; r < 8; ++r) { acc[0][g][r] = bi; acc[1][g][r] = bi; }
  }

  // Phase 1: input part (x_t@W for layer0, h1(t)@W2 for layer1)
  for (int k = 0; k < K0; k += 32) {
    v16bf a0 = load_a_frag(A0, lda0, m0,      k, lane);
    v16bf a1 = load_a_frag(A0, lda0, m0 + 16, k, lane);
#pragma unroll
    for (int g = 0; g < 4; ++g) {
      v16bf b = load_b_frag(Wt, ldk, g * H_SZ + n0, k, lane);
      acc[0][g] = __builtin_amdgcn_wmma_f32_16x16x32_bf16(
          false, a0, false, b, (short)0, acc[0][g], false, false);
      acc[1][g] = __builtin_amdgcn_wmma_f32_16x16x32_bf16(
          false, a1, false, b, (short)0, acc[1][g], false, false);
    }
    if (k + 64 < kcat)   // prefetch next K-chunk of the hot (weight) stream
      __builtin_prefetch(Wt + (size_t)(n0 + (lane & 15)) * ldk + k + 32, 0, 1);
  }
  // Phase 2: recurrent part (h@U), B columns continue at K0 offset
  for (int k = 0; k < H_SZ; k += 32) {
    v16bf a0 = load_a_frag(A1, H_SZ, m0,      k, lane);
    v16bf a1 = load_a_frag(A1, H_SZ, m0 + 16, k, lane);
#pragma unroll
    for (int g = 0; g < 4; ++g) {
      v16bf b = load_b_frag(Wt, ldk, g * H_SZ + n0, K0 + k, lane);
      acc[0][g] = __builtin_amdgcn_wmma_f32_16x16x32_bf16(
          false, a0, false, b, (short)0, acc[0][g], false, false);
      acc[1][g] = __builtin_amdgcn_wmma_f32_16x16x32_bf16(
          false, a1, false, b, (short)0, acc[1][g], false, false);
    }
    if (K0 + k + 64 < kcat)
      __builtin_prefetch(Wt + (size_t)(n0 + (lane & 15)) * ldk + K0 + k + 32, 0, 1);
  }

  // Epilogue: C/D layout -> VGPR r holds M = m0 + r + (lane>=16 ? 8 : 0), N = n0 + lane%16
  const int n = n0 + (lane & 15);
#pragma unroll
  for (int s = 0; s < 2; ++s) {
    const int mb = m0 + 16 * s + ((lane >> 4) << 3);
#pragma unroll
    for (int r = 0; r < 8; ++r) {
      int m = mb + r;
      size_t idx = (size_t)m * H_SZ + n;
      float iv = sigf(acc[s][0][r]);
      float fv = sigf(acc[s][1][r]);
      float gv = tanhf(acc[s][2][r]);
      float ov = sigf(acc[s][3][r]);
      float cn = fv * cst[idx] + iv * gv;
      cst[idx] = cn;
      float hn = ov * tanhf(cn);
      hout[idx] = (__bf16)hn;
      if (hsel != nullptr && (seqlen[m] - 1) == t) hsel[idx] = hn;
    }
  }
}

// ---------------------------------------------------------------------------
// Dense head: out[b][l] = relu(hsel[b] . Wd[:,l] + bd[l]);  B*3 = 1536 outputs
// ---------------------------------------------------------------------------
__global__ void k_dense(const float* __restrict__ Wd, const float* __restrict__ bd,
                        float* __restrict__ out) {
  int i = blockIdx.x * blockDim.x + threadIdx.x;
  if (i >= B_SZ * 3) return;
  int b = i / 3, l = i % 3;
  const float* h = g_hsel + (size_t)b * H_SZ;
  float s = bd[l];
  for (int j = 0; j < H_SZ; ++j) s += h[j] * Wd[(size_t)j * 3 + l];
  out[i] = s > 0.f ? s : 0.f;
}

// ---------------------------------------------------------------------------
extern "C" void kernel_launch(void* const* d_in, const int* in_sizes, int n_in,
                              void* d_out, int out_size, void* d_ws, size_t ws_size,
                              hipStream_t stream) {
  (void)in_sizes; (void)n_in; (void)out_size; (void)d_ws; (void)ws_size;
  const float* chars  = (const float*)d_in[0];
  const int*   seqlen = (const int*)d_in[1];
  const float* W1 = (const float*)d_in[2];
  const float* U1 = (const float*)d_in[3];
  const float* b1 = (const float*)d_in[4];
  const float* W2 = (const float*)d_in[5];
  const float* U2 = (const float*)d_in[6];
  const float* b2 = (const float*)d_in[7];
  const float* Wd = (const float*)d_in[8];
  const float* bd = (const float*)d_in[9];
  float* out = (float*)d_out;

  { int n = B_SZ * T_SZ * DIN;
    k_cvt_x<<<(n + 255) / 256, 256, 0, stream>>>(chars, n); }
  { int n = G4 * K1CAT;
    k_pack_w1<<<(n + 255) / 256, 256, 0, stream>>>(W1, U1); }
  { int n = G4 * K2CAT;
    k_pack_w2<<<(n + 255) / 256, 256, 0, stream>>>(W2, U2); }
  { int n = B_SZ * H_SZ;
    k_zero_state<<<(n + 255) / 256, 256, 0, stream>>>(); }

  dim3 grid(H_SZ / 16, B_SZ / 128);   // 64 x 4 blocks, 4 waves each
  for (int t = 0; t < T_SZ; ++t) {
    k_lstm_step<<<grid, 128, 0, stream>>>(0, t, b1, seqlen);
    k_lstm_step<<<grid, 128, 0, stream>>>(1, t, b2, seqlen);
  }

  k_dense<<<(B_SZ * 3 + 255) / 256, 256, 0, stream>>>(Wd, bd, out);
}
